// GraphSAGE_87076166959945
// MI455X (gfx1250) — matrix operand order repaired and checked by reading, not compiled, not run on hardware
//
#include <hip/hip_runtime.h>
#include <hip/hip_bf16.h>

// ---------------------------------------------------------------------------
// GraphSAGE (2-layer, mean aggregation) for MI455X / gfx1250.
//   Layer1: h   = relu([mean(x)|x] @ [W1l;W1r] + b1)   K=256, N=256
//   Layer2: out =      [mean(h)|h] @ [W2l;W2r] + b2    K=512, N=64
// GEMMs run as bf16 WMMA (v_wmma_f32_16x16x32_bf16) with f32 accumulate.
// Each wave computes a 32x64 tile: B fragments are reused across two row
// tiles (8 WMMAs per 4 b128 A loads + 64 scalar B loads per k-step).
// Scatter-mean uses coalesced float4 gathers + global_atomic_add_f32.
// ---------------------------------------------------------------------------

typedef __attribute__((ext_vector_type(16))) __bf16 v16bf;
typedef __attribute__((ext_vector_type(8)))  __bf16 v8bf;
typedef __attribute__((ext_vector_type(8)))  float  v8f;

#define N_NODES 50000
#define E_EDGES 800000

__device__ __forceinline__ __bf16 f2bf(float f) {
    // round-to-nearest-even float -> bf16
    unsigned u = __float_as_uint(f);
    unsigned r = (u + 0x7FFFu + ((u >> 16) & 1u)) >> 16;
    unsigned short s = (unsigned short)r;
    return __builtin_bit_cast(__bf16, s);
}

// ---------------- degree (in-degree of dst) --------------------------------
__global__ void deg_kernel(const int* __restrict__ dst, float* __restrict__ deg, int E) {
    int e = blockIdx.x * blockDim.x + threadIdx.x;
    if (e < E) atomicAdd(&deg[dst[e]], 1.0f);
}

// ---------------- scatter-sum: agg[dst] += feat[src] -----------------------
// Flat over E*(D/4): each thread loads one float4 of the source row and does
// 4 float atomic adds to the destination row.
template <int D>
__global__ void scatter_kernel(const int* __restrict__ src,
                               const int* __restrict__ dst,
                               const float* __restrict__ feat,
                               float* __restrict__ agg) {
    const int Q = D / 4;
    long long idx = (long long)blockIdx.x * blockDim.x + threadIdx.x;
    if (idx >= (long long)E_EDGES * Q) return;
    int e = (int)(idx / Q);
    int j = (int)(idx - (long long)e * Q) * 4;
    int s = src[e];
    int d = dst[e];
    const float4 v = *(const float4*)(feat + (size_t)s * D + j);
    float* ap = agg + (size_t)d * D + j;
    atomicAdd(ap + 0, v.x);
    atomicAdd(ap + 1, v.y);
    atomicAdd(ap + 2, v.z);
    atomicAdd(ap + 3, v.w);
}

// ---------------- stack [Wl; Wr] -> bf16 -----------------------------------
__global__ void catw_kernel(const float* __restrict__ Wl, const float* __restrict__ Wr,
                            __bf16* __restrict__ out, int rowsL, int N, int rowsTot) {
    int i = blockIdx.x * blockDim.x + threadIdx.x;
    int tot = rowsTot * N;
    if (i >= tot) return;
    int k = i / N, n = i - k * N;
    float v = (k < rowsL) ? Wl[(size_t)k * N + n] : Wr[(size_t)(k - rowsL) * N + n];
    out[i] = f2bf(v);
}

// ---------------- A1 = bf16([agg1/deg | x]) : [N_NODES x 256] --------------
__global__ void build_a1(const float* __restrict__ agg1, const float* __restrict__ x,
                         const float* __restrict__ deg, __bf16* __restrict__ A1) {
    int i = blockIdx.x * blockDim.x + threadIdx.x;
    if (i >= N_NODES * 256) return;
    int node = i >> 8, j = i & 255;
    float v;
    if (j < 128) {
        float d = deg[node]; d = d > 1.0f ? d : 1.0f;
        v = agg1[(size_t)node * 128 + j] / d;
    } else {
        v = x[(size_t)node * 128 + (j - 128)];
    }
    A1[i] = f2bf(v);
}

// ---------------- A2[:,0:256] = bf16(agg2/deg), row stride 512 -------------
__global__ void build_a2(const float* __restrict__ agg2, const float* __restrict__ deg,
                         __bf16* __restrict__ A2) {
    int i = blockIdx.x * blockDim.x + threadIdx.x;
    if (i >= N_NODES * 256) return;
    int node = i >> 8, j = i & 255;
    float d = deg[node]; d = d > 1.0f ? d : 1.0f;
    A2[(size_t)node * 512 + j] = f2bf(agg2[i] / d);
}

// ---------------- WMMA GEMM: out = A[MxK] @ W[KxN] + bias ------------------
// One 256-thread block = 8 waves; each wave owns a 32x64 tile (2 row tiles x
// 4 col tiles = 8 accumulators). B fragments are loaded once per k-step and
// reused for both row tiles.
// A-fragment layout (16-bit A 16x32): lanes 0-15 rows, K lo-halves; lanes
// 16-31 same rows, K hi-halves -> two 16B loads per lane.
// B-fragment layout (16-bit B 32x16): lane indexes N column (lane_lo), each lane
// holds 16 K values in A's k order -> 16 strided 2-byte loads per lane.
template <int K, int NTOT, bool RELU, bool EMIT_BF16>
__global__ __launch_bounds__(256)
void sage_gemm(const __bf16* __restrict__ A,
               const __bf16* __restrict__ W,
               const float*  __restrict__ bias,
               float* __restrict__ out,
               __bf16* __restrict__ out_bf, int strideBf, int offBf,
               int M) {
    const int wave    = threadIdx.x >> 5;
    const int lane    = threadIdx.x & 31;
    const int lane_lo = lane & 15;
    const int lane_hi = lane >> 4;

    const int rowBase = (blockIdx.x * 8 + wave) * 32;
    const int colBase = blockIdx.y * 64;

    int arow0 = rowBase + lane_lo;
    int arow1 = rowBase + 16 + lane_lo;
    if (arow0 > M - 1) arow0 = M - 1;        // clamp loads; stores guarded
    if (arow1 > M - 1) arow1 = M - 1;
    const __bf16* aptr0 = A + (size_t)arow0 * K;
    const __bf16* aptr1 = A + (size_t)arow1 * K;

    v8f acc[2][4] = {};

    for (int k0 = 0; k0 < K; k0 += 32) {
        // A fragments: K = k0 + hi*8 .. +7  and  k0+16+hi*8 .. +7
        v8bf a00 = *(const v8bf*)(aptr0 + k0 + lane_hi * 8);
        v8bf a01 = *(const v8bf*)(aptr0 + k0 + 16 + lane_hi * 8);
        v8bf a10 = *(const v8bf*)(aptr1 + k0 + lane_hi * 8);
        v8bf a11 = *(const v8bf*)(aptr1 + k0 + 16 + lane_hi * 8);
        v16bf af0, af1;
#pragma unroll
        for (int t = 0; t < 8; ++t) {
            af0[t] = a00[t]; af0[t + 8] = a01[t];
            af1[t] = a10[t]; af1[t + 8] = a11[t];
        }

        const __bf16* wrow = W + (size_t)(k0 + lane_hi * 8) * NTOT + colBase + lane_lo;
#pragma unroll
        for (int j = 0; j < 4; ++j) {
            v16bf bfrag;
#pragma unroll
            for (int t = 0; t < 8; ++t) {
                bfrag[t]     = wrow[(size_t)t * NTOT + j * 16];
                bfrag[t + 8] = wrow[(size_t)(16 + t) * NTOT + j * 16];
            }
            acc[0][j] = __builtin_amdgcn_wmma_f32_16x16x32_bf16(
                false, af0, false, bfrag, (short)0, acc[0][j], false, false);
            acc[1][j] = __builtin_amdgcn_wmma_f32_16x16x32_bf16(
                false, af1, false, bfrag, (short)0, acc[1][j], false, false);
        }
    }

    // Epilogue. C/D layout: VGPR r -> (M = r + lane_hi*8, N = lane_lo).
#pragma unroll
    for (int rt = 0; rt < 2; ++rt) {
        const int rowTile = rowBase + rt * 16;
#pragma unroll
        for (int j = 0; j < 4; ++j) {
            int col = colBase + j * 16 + lane_lo;
            float bv = bias[col];
#pragma unroll
            for (int r = 0; r < 8; ++r) {
                int row = rowTile + r + lane_hi * 8;
                if (row < M) {
                    float v = acc[rt][j][r] + bv;
                    if (RELU) v = v > 0.0f ? v : 0.0f;
                    out[(size_t)row * NTOT + col] = v;
                    if (EMIT_BF16)
                        out_bf[(size_t)row * strideBf + offBf + col] = f2bf(v);
                }
            }
        }
    }
}

// ---------------------------------------------------------------------------
static inline char* carve(char*& p, size_t bytes) {
    char* r = p;
    p += (bytes + 255) & ~(size_t)255;
    return r;
}

extern "C" void kernel_launch(void* const* d_in, const int* in_sizes, int n_in,
                              void* d_out, int out_size, void* d_ws, size_t ws_size,
                              hipStream_t stream) {
    const float* x   = (const float*)d_in[0];
    const int*   ei  = (const int*)d_in[1];      // edge_index [2 x E]
    const float* W1l = (const float*)d_in[2];
    const float* W1r = (const float*)d_in[3];
    const float* b1  = (const float*)d_in[4];
    const float* W2l = (const float*)d_in[5];
    const float* W2r = (const float*)d_in[6];
    const float* b2  = (const float*)d_in[7];
    float* out = (float*)d_out;

    const int* src = ei;
    const int* dst = ei + E_EDGES;

    char* p = (char*)d_ws;
    float*  deg  = (float*) carve(p, (size_t)N_NODES * 4);
    float*  agg1 = (float*) carve(p, (size_t)N_NODES * 128 * 4);
    __bf16* A1   = (__bf16*)carve(p, (size_t)N_NODES * 256 * 2);
    float*  h    = (float*) carve(p, (size_t)N_NODES * 256 * 4);
    float*  agg2 = (float*) carve(p, (size_t)N_NODES * 256 * 4);
    __bf16* A2   = (__bf16*)carve(p, (size_t)N_NODES * 512 * 2);
    __bf16* W1c  = (__bf16*)carve(p, (size_t)256 * 256 * 2);
    __bf16* W2c  = (__bf16*)carve(p, (size_t)512 * 64 * 2);

    hipMemsetAsync(deg,  0, (size_t)N_NODES * 4,        stream);
    hipMemsetAsync(agg1, 0, (size_t)N_NODES * 128 * 4,  stream);
    hipMemsetAsync(agg2, 0, (size_t)N_NODES * 256 * 4,  stream);

    // degree + layer-1 scatter-sum
    deg_kernel<<<(E_EDGES + 255) / 256, 256, 0, stream>>>(dst, deg, E_EDGES);
    {
        long long work = (long long)E_EDGES * (128 / 4);
        scatter_kernel<128><<<(unsigned)((work + 255) / 256), 256, 0, stream>>>(
            src, dst, x, agg1);
    }

    // bf16 weight stacking
    catw_kernel<<<(256 * 256 + 255) / 256, 256, 0, stream>>>(W1l, W1r, W1c, 128, 256, 256);
    catw_kernel<<<(512 * 64 + 255) / 256, 256, 0, stream>>>(W2l, W2r, W2c, 256, 64, 512);

    // A1 = [agg1/deg | x] in bf16
    build_a1<<<(N_NODES * 256 + 255) / 256, 256, 0, stream>>>(agg1, x, deg, A1);

    // Layer-1 GEMM: h = relu(A1 @ W1c + b1); also emit bf16 h into A2[:,256:512]
    {
        dim3 grid((N_NODES + 255) / 256, 256 / 64);
        sage_gemm<256, 256, true, true><<<grid, 256, 0, stream>>>(
            A1, W1c, b1, h, A2, 512, 256, N_NODES);
    }

    // layer-2 scatter-sum on h
    {
        long long work = (long long)E_EDGES * (256 / 4);
        scatter_kernel<256><<<(unsigned)((work + 255) / 256), 256, 0, stream>>>(
            src, dst, h, agg2);
    }

    // A2[:,0:256] = bf16(agg2/deg)
    build_a2<<<(N_NODES * 256 + 255) / 256, 256, 0, stream>>>(agg2, deg, A2);

    // Layer-2 GEMM: out = A2 @ W2c + b2
    {
        dim3 grid((N_NODES + 255) / 256, 1);
        sage_gemm<512, 64, false, false><<<grid, 256, 0, stream>>>(
            A2, W2c, b2, out, (__bf16*)nullptr, 0, 0, N_NODES);
    }
}
